// MultiHeadCrossAttention_43559558316708
// MI455X (gfx1250) — matrix operand order repaired and checked by reading, not compiled, not run on hardware
//
#include <hip/hip_runtime.h>

// ---------------------------------------------------------------------------
// MultiHeadCrossAttention for MI455X (gfx1250, wave32, WMMA).
// All matmuls (Q/K/V proj, QK^T, PV, out proj) use v_wmma_f32_16x16x32_bf16.
// ---------------------------------------------------------------------------

typedef __attribute__((ext_vector_type(16))) __bf16 v16bf;
typedef __attribute__((ext_vector_type(8)))  float  v8f;
typedef unsigned short bfb;   // bf16 bit pattern
typedef unsigned int   u32;

#define EMB  256
#define NROW 4096
#define HEADS 4
#define HDIM 64

union TileU {
    v16bf v;
    uint4 q[2];
};

static __device__ __forceinline__ v8f vzero8() { v8f z = {}; return z; }

// float -> bf16 bits, round-to-nearest-even
static __device__ __forceinline__ bfb f2bf(float f) {
    u32 u = __float_as_uint(f);
    u32 lsb = (u >> 16) & 1u;
    u += 0x7FFFu + lsb;
    return (bfb)(u >> 16);
}

// A-operand tile (16 rows x 32 K) from a row-major bf16 matrix.
// ISA layout (CDNA5 ISA, 16-bit A 16x32): lane L -> row = L&15, half h = L>>4;
// elements 0..7 hold K = h*8..h*8+7, elements 8..15 hold K = 16+h*8..16+h*8+7.
// Both element groups are 16B-contiguous -> two global_load_b128 per lane.
static __device__ __forceinline__ v16bf load_tile_a(const bfb* __restrict__ p,
                                                    int ld, int row0, int col0) {
    int lane = threadIdx.x & 31;
    int m = lane & 15;
    int h = lane >> 4;
    const bfb* base = p + (size_t)(row0 + m) * ld + col0 + h * 8;
    TileU t;
    t.q[0] = *(const uint4*)(base);        // K = h*8 + 0..7
    t.q[1] = *(const uint4*)(base + 16);   // K = 16 + h*8 + 0..7
    return t.v;
}

// B-operand tile (32 K x 16 cols) whose column n comes from row (n0+n) of a
// row-major source (i.e. computing X * S^T with S rows as output columns).
// ISA layout (CDNA5 ISA B-matrix): lane L -> col n = L&15, group g = L>>4;
// elements 0..7 hold K = g*8 + 0..7, elements 8..15 hold K = 16 + g*8 + 0..7.
// Two 16B-contiguous runs per lane -> two global_load_b128.
static __device__ __forceinline__ v16bf load_tile_b(const bfb* __restrict__ p,
                                                    int ld, int n0, int k0) {
    int lane = threadIdx.x & 31;
    int n = lane & 15;
    int g = lane >> 4;
    const bfb* base = p + (size_t)(n0 + n) * ld + k0 + g * 8;
    TileU t;
    t.q[0] = *(const uint4*)(base);        // K = g*8 + 0..7
    t.q[1] = *(const uint4*)(base + 16);   // K = 16 + g*8 + 0..7
    return t.v;
}

static __device__ __forceinline__ v8f wmma_bf16(v16bf a, v16bf b, v8f c) {
    return __builtin_amdgcn_wmma_f32_16x16x32_bf16(false, a, false, b,
                                                   (short)0, c, false, false);
}

// ---------------------------------------------------------------------------
// f32 -> bf16 convert
// ---------------------------------------------------------------------------
__global__ void cvt_f32_bf16(const float* __restrict__ in, bfb* __restrict__ out, int n) {
    int i = blockIdx.x * blockDim.x + threadIdx.x;
    if (i < n) out[i] = f2bf(in[i]);
}

// ---------------------------------------------------------------------------
// Y[M,256] = alpha * X[M,256] @ W[256,256]^T  (+ bias + residual for mode 2)
// mode 0: bf16 out, row-major [M,256]
// mode 1: bf16 out, transposed [256,M]   (used for V so PV loads are contiguous)
// mode 2: f32 out  [M,256], += bias[col] + resid[row,col]
// One wave per 16x16 tile, 8 waves per block. K = 256 -> 8 WMMAs.
// ---------------------------------------------------------------------------
__global__ __launch_bounds__(256)
void gemm_nt(const bfb* __restrict__ Xb, const bfb* __restrict__ Wb,
             bfb* __restrict__ Ob, float* __restrict__ Of,
             const float* __restrict__ bias, const float* __restrict__ resid,
             int M, int mode, float alpha) {
    int wave = threadIdx.x >> 5;
    int lane = threadIdx.x & 31;
    int r0 = (blockIdx.x * 8 + wave) * 16;
    int c0 = blockIdx.y * 16;
    if (r0 >= M) return;

    v8f acc = vzero8();
#pragma unroll
    for (int k = 0; k < EMB; k += 32) {
        v16bf a = load_tile_a(Xb, EMB, r0, k);   // X rows
        v16bf b = load_tile_b(Wb, EMB, c0, k);   // W rows = output cols
        acc = wmma_bf16(a, b, acc);
    }

    int g = lane >> 4;          // row group: rows 0..7 / 8..15
    int n = lane & 15;          // output column within tile
    if (mode == 2) {
#pragma unroll
        for (int r = 0; r < 8; ++r) {
            int row = r0 + r + 8 * g, col = c0 + n;
            Of[(size_t)row * EMB + col] =
                acc[r] * alpha + bias[col] + resid[(size_t)row * EMB + col];
        }
    } else if (mode == 1) {
#pragma unroll
        for (int r = 0; r < 8; ++r) {
            int row = r0 + r + 8 * g, col = c0 + n;
            Ob[(size_t)col * M + row] = f2bf(acc[r] * alpha);
        }
    } else {
#pragma unroll
        for (int r = 0; r < 8; ++r) {
            int row = r0 + r + 8 * g, col = c0 + n;
            Ob[(size_t)row * EMB + col] = f2bf(acc[r] * alpha);
        }
    }
}

// ---------------------------------------------------------------------------
// Flash-attention core: one wave handles (head, 16-query block), streaming
// keys in chunks of 32 with online softmax.
//   Q  : [N,256] bf16, already scaled by 1/sqrt(HDIM)
//   K  : [M,256] bf16 row-major
//   Vt : [256,M] bf16 (transposed V) so the B-operand of PV is contiguous
//   Ctx: [N,256] bf16 output
// ---------------------------------------------------------------------------
__global__ __launch_bounds__(32)
void flash16(const bfb* __restrict__ Q, const bfb* __restrict__ K,
             const bfb* __restrict__ Vt, bfb* __restrict__ Ctx,
             int N, int M) {
    __shared__ __align__(16) bfb Plds[16 * 32];   // P tile bounce buffer

    int lane = threadIdx.x & 31;
    int g    = lane >> 4;
    int nn   = lane & 15;
    int r0   = blockIdx.x * 16;          // query block
    int d0   = blockIdx.y * HDIM;        // head offset in EMB

    // Q operand tiles for K-dims 0..31 and 32..63 (resident for whole loop)
    v16bf qa0 = load_tile_a(Q, EMB, r0, d0);
    v16bf qa1 = load_tile_a(Q, EMB, r0, d0 + 32);

    float mrow[8], lrow[8];
    v8f o0 = vzero8(), o1 = vzero8(), o2 = vzero8(), o3 = vzero8();
#pragma unroll
    for (int r = 0; r < 8; ++r) { mrow[r] = -1e30f; lrow[r] = 0.f; }

    for (int kb = 0; kb < M; kb += 32) {
        if (kb + 32 < M) {
            // Pull next key/value chunk toward the caches (global_prefetch_b8)
            __builtin_prefetch(K + (size_t)(kb + 32 + nn) * EMB + d0, 0, 1);
            __builtin_prefetch(Vt + (size_t)(d0 + nn) * M + kb + 32, 0, 1);
        }
        // S = Q K^T for two 16-key sub-tiles (each over K-dim 64 -> 2 WMMAs)
        // B-operand column n = key row kb(+16) + n of K, K-dim = head dims.
        v16bf b00 = load_tile_b(K, EMB, kb,      d0);
        v16bf b01 = load_tile_b(K, EMB, kb,      d0 + 32);
        v16bf b10 = load_tile_b(K, EMB, kb + 16, d0);
        v16bf b11 = load_tile_b(K, EMB, kb + 16, d0 + 32);
        v8f s0 = wmma_bf16(qa0, b00, vzero8());
        s0     = wmma_bf16(qa1, b01, s0);
        v8f s1 = wmma_bf16(qa0, b10, vzero8());
        s1     = wmma_bf16(qa1, b11, s1);

        // Online softmax. C-layout: slot r in lane L holds row m = r + 8*(L/16),
        // key column = (L&15) (+16 for s1) -> row reduce = xor-shuffles over the
        // 16-lane half (masks 1,2,4,8 never cross bit 4).
#pragma unroll
        for (int r = 0; r < 8; ++r) {
            float t = fmaxf(s0[r], s1[r]);
            t = fmaxf(t, __shfl_xor(t, 1, 32));
            t = fmaxf(t, __shfl_xor(t, 2, 32));
            t = fmaxf(t, __shfl_xor(t, 4, 32));
            t = fmaxf(t, __shfl_xor(t, 8, 32));
            float mnew = fmaxf(mrow[r], t);
            float corr = __expf(mrow[r] - mnew);
            mrow[r] = mnew;
            float p0 = __expf(s0[r] - mnew);
            float p1 = __expf(s1[r] - mnew);
            float rs = p0 + p1;
            rs += __shfl_xor(rs, 1, 32);
            rs += __shfl_xor(rs, 2, 32);
            rs += __shfl_xor(rs, 4, 32);
            rs += __shfl_xor(rs, 8, 32);
            lrow[r] = lrow[r] * corr + rs;
            o0[r] *= corr; o1[r] *= corr; o2[r] *= corr; o3[r] *= corr;
            // scatter P (bf16) into its [m][k] position for re-layout
            int m = r + 8 * g;
            Plds[m * 32 + nn]      = f2bf(p0);
            Plds[m * 32 + 16 + nn] = f2bf(p1);
        }
        __syncthreads();   // single-wave block: effectively a ds-wait

        // Re-load P in A-operand layout: lane L -> row L&15, K halves per L>>4.
        TileU pt;
        {
            int m = lane & 15;
            const bfb* pb = Plds + m * 32 + g * 8;
            pt.q[0] = *(const uint4*)(pb);
            pt.q[1] = *(const uint4*)(pb + 16);
        }
        v16bf pa = pt.v;
        __syncthreads();   // protect Plds before next iteration's stores

        // O += P @ V : four 16-wide d sub-tiles.
        // B-operand column n = Vt row d0+ds*16+n, K-dim = keys kb..kb+31.
#pragma unroll
        for (int ds = 0; ds < 4; ++ds) {
            v16bf vb = load_tile_b(Vt, M, d0 + ds * 16, kb);
            v8f* op = (ds == 0) ? &o0 : (ds == 1) ? &o1 : (ds == 2) ? &o2 : &o3;
            *op = wmma_bf16(pa, vb, *op);
        }
    }

    // Epilogue: divide by softmax denom, write ctx in bf16
#pragma unroll
    for (int ds = 0; ds < 4; ++ds) {
        v8f oc = (ds == 0) ? o0 : (ds == 1) ? o1 : (ds == 2) ? o2 : o3;
#pragma unroll
        for (int r = 0; r < 8; ++r) {
            int row = r0 + r + 8 * g;
            float v = oc[r] / lrow[r];
            Ctx[(size_t)row * EMB + d0 + ds * 16 + nn] = f2bf(v);
        }
    }
}

// ---------------------------------------------------------------------------
// Host-side orchestration
// ---------------------------------------------------------------------------
extern "C" void kernel_launch(void* const* d_in, const int* in_sizes, int n_in,
                              void* d_out, int out_size, void* d_ws, size_t ws_size,
                              hipStream_t stream) {
    (void)in_sizes; (void)n_in; (void)out_size; (void)ws_size;
    const float* A  = (const float*)d_in[0];
    const float* B  = (const float*)d_in[1];
    const float* Wq = (const float*)d_in[2];
    const float* Wk = (const float*)d_in[3];
    const float* Wv = (const float*)d_in[4];
    const float* Wo = (const float*)d_in[5];
    const float* bo = (const float*)d_in[6];
    float* Out = (float*)d_out;

    // carve bf16 scratch out of d_ws (~20.5 MiB total)
    char* ws = (char*)d_ws;
    size_t off = 0;
    auto carve = [&](size_t elems) {
        bfb* p = (bfb*)(ws + off);
        off += (elems * sizeof(bfb) + 255) & ~(size_t)255;
        return p;
    };
    bfb* Abf  = carve((size_t)NROW * EMB);
    bfb* Bbf  = carve((size_t)NROW * EMB);
    bfb* Wqb  = carve((size_t)EMB * EMB);
    bfb* Wkb  = carve((size_t)EMB * EMB);
    bfb* Wvb  = carve((size_t)EMB * EMB);
    bfb* Wob  = carve((size_t)EMB * EMB);
    bfb* QA   = carve((size_t)NROW * EMB);
    bfb* KA   = carve((size_t)NROW * EMB);
    bfb* VtA  = carve((size_t)NROW * EMB);
    bfb* QB   = carve((size_t)NROW * EMB);
    bfb* KB   = carve((size_t)NROW * EMB);
    bfb* VtB  = carve((size_t)NROW * EMB);
    bfb* CtxA = carve((size_t)NROW * EMB);
    bfb* CtxB = carve((size_t)NROW * EMB);

    const int nAct = NROW * EMB;   // 1,048,576
    const int nW   = EMB * EMB;    // 65,536
    cvt_f32_bf16<<<(nAct + 255) / 256, 256, 0, stream>>>(A,  Abf, nAct);
    cvt_f32_bf16<<<(nAct + 255) / 256, 256, 0, stream>>>(B,  Bbf, nAct);
    cvt_f32_bf16<<<(nW   + 255) / 256, 256, 0, stream>>>(Wq, Wqb, nW);
    cvt_f32_bf16<<<(nW   + 255) / 256, 256, 0, stream>>>(Wk, Wkb, nW);
    cvt_f32_bf16<<<(nW   + 255) / 256, 256, 0, stream>>>(Wv, Wvb, nW);
    cvt_f32_bf16<<<(nW   + 255) / 256, 256, 0, stream>>>(Wo, Wob, nW);

    dim3 gg(NROW / 128, EMB / 16);   // 32 x 16 blocks, 8 waves each
    const float qscale = 0.125f;     // 1/sqrt(HDIM)
    // Projections (Q pre-scaled, V stored transposed)
    gemm_nt<<<gg, 256, 0, stream>>>(Abf, Wqb, QA,  nullptr, nullptr, nullptr, NROW, 0, qscale);
    gemm_nt<<<gg, 256, 0, stream>>>(Abf, Wkb, KA,  nullptr, nullptr, nullptr, NROW, 0, 1.0f);
    gemm_nt<<<gg, 256, 0, stream>>>(Abf, Wvb, VtA, nullptr, nullptr, nullptr, NROW, 1, 1.0f);
    gemm_nt<<<gg, 256, 0, stream>>>(Bbf, Wqb, QB,  nullptr, nullptr, nullptr, NROW, 0, qscale);
    gemm_nt<<<gg, 256, 0, stream>>>(Bbf, Wkb, KB,  nullptr, nullptr, nullptr, NROW, 0, 1.0f);
    gemm_nt<<<gg, 256, 0, stream>>>(Bbf, Wvb, VtB, nullptr, nullptr, nullptr, NROW, 1, 1.0f);

    // Attention: A attends to B, B attends to A
    flash16<<<dim3(NROW / 16, HEADS), 32, 0, stream>>>(QA, KB, VtB, CtxA, NROW, NROW);
    flash16<<<dim3(NROW / 16, HEADS), 32, 0, stream>>>(QB, KA, VtA, CtxB, NROW, NROW);

    // Output projection + bias + residual, f32 into d_out (A_out then B_out)
    gemm_nt<<<gg, 256, 0, stream>>>(CtxA, Wob, nullptr, Out,                     bo, A, NROW, 2, 1.0f);
    gemm_nt<<<gg, 256, 0, stream>>>(CtxB, Wob, nullptr, Out + (size_t)NROW * EMB, bo, B, NROW, 2, 1.0f);
}
